// SelfAttentionFusion_18081812316914
// MI455X (gfx1250) — matrix operand order repaired and checked by reading, not compiled, not run on hardware
//
#include <hip/hip_runtime.h>
#include <hip/hip_bf16.h>

typedef __attribute__((ext_vector_type(16))) __bf16        v16bf;
typedef __attribute__((ext_vector_type(8)))  float         v8f;
typedef __attribute__((ext_vector_type(8)))  unsigned int  v8u;
typedef __attribute__((ext_vector_type(4)))  unsigned int  v4u;
typedef __attribute__((ext_vector_type(8)))  int           v8i;
typedef __attribute__((ext_vector_type(4)))  int           v4i;

__device__ __forceinline__ unsigned short f32_to_bf16(float f) {
    union { float f; unsigned int u; } v; v.f = f;
    unsigned int r = v.u + 0x7FFFu + ((v.u >> 16) & 1u);   // round-to-nearest-even
    return (unsigned short)(r >> 16);
}

__device__ __forceinline__ v8f wmma_bf16(v16bf a, v16bf b, v8f c) {
    return __builtin_amdgcn_wmma_f32_16x16x32_bf16(false, a, false, b, (short)0, c,
                                                   false, false);
}

// ---------------------------------------------------------------------------
// Tensor Data Mover: 2-D tile load (global -> LDS).  D# built per CDNA5 ISA
// CDNA5 ISA: group0 = {flags, lds_addr, global_addr, type=2},
// group1 = {data_size=2B, pad cfg, tensor dims/strides, tile dims}.
// dim0 = contiguous axis.  pad_interval code: 3 -> every 16 DWORDs (64B rows),
// 6 -> every 128 DWORDs (512B rows); pad_amount code 0 -> 1 DWORD.
// ---------------------------------------------------------------------------
__device__ __forceinline__ void tdm_load_2d(unsigned lds_off, const void* gptr,
                                            unsigned td0, unsigned td1,
                                            unsigned stride0,
                                            unsigned tile0, unsigned tile1,
                                            unsigned pad_int, unsigned pad_amt) {
    unsigned long long ga = (unsigned long long)(uintptr_t)gptr;
    v4u g0;
    g0[0] = 1u;                                   // count=1 (valid), normal mode
    g0[1] = lds_off;                              // lds_addr (bytes)
    g0[2] = (unsigned)ga;                         // global_addr[31:0]
    g0[3] = ((unsigned)(ga >> 32) & 0x01FFFFFFu)  // global_addr[56:32]
            | (2u << 30);                         // type = 2 ("image")
    v8i g1;
    g1[0] = (int)((1u << 16)                      // data_size = 2 bytes
                  | (1u << 20)                    // pad_enable
                  | (pad_int << 22) | (pad_amt << 25));
    g1[1] = (int)(td0 << 16);                     // tensor_dim0 low16 @ bits 63:48
    g1[2] = (int)((td0 >> 16) | (td1 << 16));     // tensor_dim0 hi / dim1 lo
    g1[3] = (int)((td1 >> 16) | (tile0 << 16));   // tensor_dim1 hi / tile_dim0
    g1[4] = (int)(tile1 & 0xFFFFu);               // tile_dim1 (tile_dim2 = 0)
    g1[5] = (int)stride0;                         // tensor_dim0_stride low32
    g1[6] = 0;                                    // stride0 hi16 / stride1 lo16
    g1[7] = 0;
    v4i gz = {0, 0, 0, 0};
#if defined(__clang_major__) && (__clang_major__ >= 23)
    v8i gz8 = {0, 0, 0, 0, 0, 0, 0, 0};
    __builtin_amdgcn_tensor_load_to_lds(g0, g1, gz, gz, gz8, 0);
#else
    __builtin_amdgcn_tensor_load_to_lds(g0, g1, gz, gz, 0);
#endif
}

// ---------------------------------------------------------------------------
// Kernel A: q/k/v = W x + b, fused with vit_flatten.
// fq/fk: [16 bb][2048 t][256 n] bf16 ; fv: TRANSPOSED [16 bb][256 n][2048 t]
// so the attention kernel can TDM-load V chunks directly.
// ---------------------------------------------------------------------------
__global__ __launch_bounds__(256)
void qkv_proj(const float* __restrict__ rgb, const float* __restrict__ depth,
              const float* __restrict__ wq1, const float* __restrict__ bq1,
              const float* __restrict__ wk1, const float* __restrict__ bk1,
              const float* __restrict__ wv1, const float* __restrict__ bv1,
              const float* __restrict__ wq2, const float* __restrict__ bq2,
              const float* __restrict__ wk2, const float* __restrict__ bk2,
              const float* __restrict__ wv2, const float* __restrict__ bv2,
              unsigned short* __restrict__ fq, unsigned short* __restrict__ fk,
              unsigned short* __restrict__ fv) {
    const int bb = blockIdx.y;           // b*2 + branch
    const int b = bb >> 1, br = bb & 1;
    const float* x  = (br ? depth : rgb) + (size_t)b * 512 * 16384;
    const float* wq = br ? wq2 : wq1;  const float* bq = br ? bq2 : bq1;
    const float* wk = br ? wk2 : wk1;  const float* bk = br ? bk2 : bk1;
    const float* wv = br ? wv2 : wv1;  const float* bv = br ? bv2 : bv1;

    const int tid = threadIdx.x;
    const int wave = tid >> 5, lane = tid & 31;
    const int half = lane >> 4, lm = lane & 15;
    const int mt = wave & 1, nt = wave >> 1;   // 2 M-tiles x 4 N-tiles
    const int pixBase = blockIdx.x * 64;

    __shared__ __attribute__((aligned(16))) unsigned short xs[64][34];
    __shared__ __attribute__((aligned(16))) unsigned short wsh[3][32][34];

    v8f cq = {}; v8f ck = {}; v8f cv = {};

    for (int k0 = 0; k0 < 512; k0 += 32) {
        __syncthreads();
        #pragma unroll
        for (int i = 0; i < 8; ++i) {
            int e = tid + i * 256;
            int c = e >> 6, p = e & 63;
            xs[p][c] = f32_to_bf16(x[(size_t)(k0 + c) * 16384 + pixBase + p]);
        }
        if (k0 + 32 < 512)   // prefetch next x tile (-> global_prefetch_b8)
            __builtin_prefetch(&x[(size_t)(k0 + 32 + (tid >> 6)) * 16384 +
                                  pixBase + (tid & 63)], 0, 0);
        #pragma unroll
        for (int i = 0; i < 12; ++i) {
            int e = tid + i * 256;
            int mat = e >> 10, r = e & 1023;
            int o = r >> 5, c = r & 31;
            const float* wp = (mat == 0) ? wq : (mat == 1 ? wk : wv);
            wsh[mat][o][c] = f32_to_bf16(wp[o * 512 + k0 + c]);
        }
        __syncthreads();

        v8u bx, pq, pk, pv;
        #pragma unroll
        for (int v = 0; v < 8; ++v) {
            int kb = half * 16 + 2 * v;                         // B frag K index
            int ka = half * 16 + 2 * v;                         // A frag K index
            bx[v] = *(const unsigned int*)&xs[nt * 16 + lm][kb];
            pq[v] = *(const unsigned int*)&wsh[0][mt * 16 + lm][ka];
            pk[v] = *(const unsigned int*)&wsh[1][mt * 16 + lm][ka];
            pv[v] = *(const unsigned int*)&wsh[2][mt * 16 + lm][ka];
        }
        v16bf bxf = __builtin_bit_cast(v16bf, bx);
        cq = wmma_bf16(__builtin_bit_cast(v16bf, pq), bxf, cq);
        ck = wmma_bf16(__builtin_bit_cast(v16bf, pk), bxf, ck);
        cv = wmma_bf16(__builtin_bit_cast(v16bf, pv), bxf, cv);
    }

    const int pix = pixBase + nt * 16 + lm;
    const int h = pix >> 7, w = pix & 127;
    const int hp = h >> 3, py = h & 7, wp2 = w >> 3, px = w & 7;
    const int n = hp * 16 + wp2;
    const size_t base = (size_t)bb * 2048 * 256;
    #pragma unroll
    for (int v = 0; v < 8; ++v) {
        int o = mt * 16 + v + 8 * half;          // C layout: VGPR r -> M=r+8*half
        int t = o * 64 + py * 8 + px;
        fq[base + (size_t)t * 256 + n]  = f32_to_bf16(cq[v] + bq[o]);
        fk[base + (size_t)t * 256 + n]  = f32_to_bf16(ck[v] + bk[o]);
        fv[base + (size_t)n * 2048 + t] = f32_to_bf16(cv[v] + bv[o]);  // transposed
    }
}

// ---------------------------------------------------------------------------
// Kernel B: flash attention, 2048 tokens x 256 dim, per (batch,branch).
// K and V chunks are DMA'd into LDS by the Tensor Data Mover, double-buffered
// so the DMA of chunk i+1 overlaps the WMMA work on chunk i.
// ---------------------------------------------------------------------------
__global__ __launch_bounds__(256)
void flash_attn(const unsigned short* __restrict__ fq,
                const unsigned short* __restrict__ fk,
                const unsigned short* __restrict__ fv_t,
                unsigned short* __restrict__ fo) {
    const int bb = blockIdx.y;
    const int tid = threadIdx.x, wave = tid >> 5, lane = tid & 31;
    const int half = lane >> 4, lm = lane & 15;
    const int t0 = blockIdx.x * 128 + wave * 16;     // 16 query rows per wave
    const size_t tb = (size_t)bb * 2048 * 256;

    // TDM pads 1 DWORD per row: K rows 256 elem -> pitch 258, V rows 32 -> 34.
    __shared__ __attribute__((aligned(16))) unsigned short klds[2][32][258];
    __shared__ __attribute__((aligned(16))) unsigned short vlds[2][256][34];
    __shared__ __attribute__((aligned(16))) unsigned short asmem[8][16][32];

    const unsigned int* fq32 = (const unsigned int*)fq;
    const unsigned short* fkbb = fk + tb;     // [2048 s][256 n]
    const unsigned short* fvbb = fv_t + tb;   // [256 n][2048 s]

    // preload Q fragments (8 K-steps over the 256-feature dim)
    v16bf qf[8];
    #pragma unroll
    for (int st = 0; st < 8; ++st) {
        v8u p;
        #pragma unroll
        for (int v = 0; v < 8; ++v) {
            int kk = st * 32 + half * 16 + 2 * v;
            p[v] = fq32[(tb + (size_t)(t0 + lm) * 256 + kk) >> 1];
        }
        qf[st] = __builtin_bit_cast(v16bf, p);
    }

    float rmax[8], rsum[8];
    #pragma unroll
    for (int v = 0; v < 8; ++v) { rmax[v] = -3.4e38f; rsum[v] = 0.f; }
    v8f acc[16];
    const v8f vzero = {};
    #pragma unroll
    for (int i = 0; i < 16; ++i) acc[i] = vzero;

    const float scale = 0.022097086912079608f;       // 1/sqrt(2048)

    // prologue: DMA chunk 0 into buffer 0
    if (wave == 0) {
        tdm_load_2d((unsigned)(uintptr_t)&klds[0][0][0], fkbb,
                    256, 2048, 256, 256, 32, 6, 0);   // K: dim0=n contiguous
        tdm_load_2d((unsigned)(uintptr_t)&vlds[0][0][0], fvbb,
                    2048, 256, 2048, 32, 256, 3, 0);  // V: dim0=s contiguous
    }

    for (int it = 0; it < 64; ++it) {
        const int s0 = it * 32;
        const int buf = it & 1;
        __syncthreads();   // everyone done reading buf^1 (iteration it-1)
        if (wave == 0) {
            if (it + 1 < 64) {
                tdm_load_2d((unsigned)(uintptr_t)&klds[buf ^ 1][0][0],
                            fkbb + (size_t)(s0 + 32) * 256,
                            256, 2048, 256, 256, 32, 6, 0);
                tdm_load_2d((unsigned)(uintptr_t)&vlds[buf ^ 1][0][0],
                            fvbb + (s0 + 32),
                            2048, 256, 2048, 32, 256, 3, 0);
                __builtin_amdgcn_s_wait_tensorcnt((short)2);  // chunk it complete
            } else {
                __builtin_amdgcn_s_wait_tensorcnt((short)0);
            }
        }
        __syncthreads();   // buffer `buf` holds chunk it for all waves

        // scores: Q[16x256] . K^T[256x32] -> two 16x16 tiles
        v8f S0 = {}; v8f S1 = {};
        #pragma unroll
        for (int st = 0; st < 8; ++st) {
            int n0 = st * 32;
            v8u p0, p1;
            #pragma unroll
            for (int v = 0; v < 8; ++v) {
                int kk = n0 + half * 16 + 2 * v;
                p0[v] = *(const unsigned int*)&klds[buf][lm][kk];
                p1[v] = *(const unsigned int*)&klds[buf][16 + lm][kk];
            }
            S0 = wmma_bf16(qf[st], __builtin_bit_cast(v16bf, p0), S0);
            S1 = wmma_bf16(qf[st], __builtin_bit_cast(v16bf, p1), S1);
        }
        S0 = S0 * scale;  S1 = S1 * scale;

        // online softmax; row stats span the 16 lanes of each half-wave
        float corr[8];
        #pragma unroll
        for (int v = 0; v < 8; ++v) {
            float tm = fmaxf(S0[v], S1[v]);
            tm = fmaxf(tm, __shfl_xor(tm, 1));
            tm = fmaxf(tm, __shfl_xor(tm, 2));
            tm = fmaxf(tm, __shfl_xor(tm, 4));
            tm = fmaxf(tm, __shfl_xor(tm, 8));
            float nm = fmaxf(rmax[v], tm);
            corr[v] = __expf(rmax[v] - nm);
            rmax[v] = nm;
            float p0 = __expf(S0[v] - nm);
            float p1 = __expf(S1[v] - nm);
            float ps = p0 + p1;
            ps += __shfl_xor(ps, 1);
            ps += __shfl_xor(ps, 2);
            ps += __shfl_xor(ps, 4);
            ps += __shfl_xor(ps, 8);
            rsum[v] = rsum[v] * corr[v] + ps;
            int row = v + 8 * half;
            asmem[wave][row][lm]      = f32_to_bf16(p0);
            asmem[wave][row][lm + 16] = f32_to_bf16(p1);
        }
        #pragma unroll
        for (int nt = 0; nt < 16; ++nt)
            #pragma unroll
            for (int v = 0; v < 8; ++v) acc[nt][v] *= corr[v];

        // P . V (A frag via wave-private LDS; same-wave DS ops are in order)
        v8u pa;
        #pragma unroll
        for (int v = 0; v < 8; ++v) {
            int kk = half * 16 + 2 * v;
            pa[v] = *(const unsigned int*)&asmem[wave][lm][kk];
        }
        v16bf af = __builtin_bit_cast(v16bf, pa);
        #pragma unroll
        for (int nt = 0; nt < 16; ++nt) {
            v8u pv;
            #pragma unroll
            for (int v = 0; v < 8; ++v)
                pv[v] = *(const unsigned int*)&vlds[buf][nt * 16 + lm]
                                                   [half * 16 + 2 * v];
            acc[nt] = wmma_bf16(af, __builtin_bit_cast(v16bf, pv), acc[nt]);
        }
    }

    // normalize + scatter to pixel layout (fused pixel_shuffle)
    float rinv[8];
    #pragma unroll
    for (int v = 0; v < 8; ++v) rinv[v] = 1.0f / rsum[v];
    #pragma unroll
    for (int nt = 0; nt < 16; ++nt) {
        int n = nt * 16 + lm;
        int hp = n >> 4, wp = n & 15;
        #pragma unroll
        for (int v = 0; v < 8; ++v) {
            int t = t0 + v + 8 * half;
            int ci = t >> 6, py = (t >> 3) & 7, px = t & 7;
            int h = hp * 8 + py, w = wp * 8 + px;
            fo[(size_t)bb * 32 * 16384 + (size_t)ci * 16384 + h * 128 + w] =
                f32_to_bf16(acc[nt][v] * rinv[v]);
        }
    }
}

// ---------------------------------------------------------------------------
// Kernel C: out = wu1 . o_rgb + bu1 + wu2 . o_depth + bu2   (K = 32)
// ---------------------------------------------------------------------------
__global__ __launch_bounds__(256)
void out_proj(const unsigned short* __restrict__ fo,
              const float* __restrict__ wu1, const float* __restrict__ bu1,
              const float* __restrict__ wu2, const float* __restrict__ bu2,
              float* __restrict__ out) {
    const int b = blockIdx.y;
    const int pixBase = blockIdx.x * 16;
    const int tid = threadIdx.x, wave = tid >> 5, lane = tid & 31;
    const int half = lane >> 4, lm = lane & 15;

    __shared__ __attribute__((aligned(16))) unsigned short osh[2][16][34];

    #pragma unroll
    for (int i = 0; i < 4; ++i) {
        int e = tid + i * 256;
        int br = e >> 9, r = e & 511;
        int ci = r >> 4, p = r & 15;
        osh[br][p][ci] = fo[(size_t)(b * 2 + br) * 32 * 16384 +
                            (size_t)ci * 16384 + pixBase + p];
    }
    __syncthreads();

    v8u q0, q1;
    #pragma unroll
    for (int v = 0; v < 8; ++v) {
        int kk = half * 16 + 2 * v;
        q0[v] = *(const unsigned int*)&osh[0][lm][kk];
        q1[v] = *(const unsigned int*)&osh[1][lm][kk];
    }
    v16bf bf0 = __builtin_bit_cast(v16bf, q0);
    v16bf bf1 = __builtin_bit_cast(v16bf, q1);

    #pragma unroll
    for (int t = 0; t < 4; ++t) {                   // 8 waves x 4 tiles = 512 ch
        int coBase = wave * 64 + t * 16;
        v8u a0, a1;
        #pragma unroll
        for (int v = 0; v < 8; ++v) {
            int kk = half * 16 + 2 * v;
            int co = coBase + lm;
            a0[v] = (unsigned int)f32_to_bf16(wu1[co * 32 + kk]) |
                    ((unsigned int)f32_to_bf16(wu1[co * 32 + kk + 1]) << 16);
            a1[v] = (unsigned int)f32_to_bf16(wu2[co * 32 + kk]) |
                    ((unsigned int)f32_to_bf16(wu2[co * 32 + kk + 1]) << 16);
        }
        v8f acc = {};
        acc = wmma_bf16(__builtin_bit_cast(v16bf, a0), bf0, acc);
        acc = wmma_bf16(__builtin_bit_cast(v16bf, a1), bf1, acc);
        #pragma unroll
        for (int v = 0; v < 8; ++v) {
            int co = coBase + v + 8 * half;
            out[((size_t)b * 512 + co) * 16384 + pixBase + lm] =
                acc[v] + bu1[co] + bu2[co];
        }
    }
}

extern "C" void kernel_launch(void* const* d_in, const int* in_sizes, int n_in,
                              void* d_out, int out_size, void* d_ws, size_t ws_size,
                              hipStream_t stream) {
    const float* rgb   = (const float*)d_in[0];
    const float* depth = (const float*)d_in[1];
    const float* wq1 = (const float*)d_in[2];  const float* bq1 = (const float*)d_in[3];
    const float* wk1 = (const float*)d_in[4];  const float* bk1 = (const float*)d_in[5];
    const float* wv1 = (const float*)d_in[6];  const float* bv1 = (const float*)d_in[7];
    const float* wq2 = (const float*)d_in[8];  const float* bq2 = (const float*)d_in[9];
    const float* wk2 = (const float*)d_in[10]; const float* bk2 = (const float*)d_in[11];
    const float* wv2 = (const float*)d_in[12]; const float* bv2 = (const float*)d_in[13];
    const float* wu1 = (const float*)d_in[14]; const float* bu1 = (const float*)d_in[15];
    const float* wu2 = (const float*)d_in[16]; const float* bu2 = (const float*)d_in[17];
    float* out = (float*)d_out;

    // workspace: fq/fk (token layout), fv (transposed), fo (pixel layout); 64 MB
    unsigned short* fq = (unsigned short*)d_ws;
    unsigned short* fk = fq + (size_t)16 * 2048 * 256;
    unsigned short* fv = fk + (size_t)16 * 2048 * 256;
    unsigned short* fo = fv + (size_t)16 * 2048 * 256;

    qkv_proj<<<dim3(256, 16), 256, 0, stream>>>(rgb, depth,
        wq1, bq1, wk1, bk1, wv1, bv1, wq2, bq2, wk2, bk2, wv2, bv2, fq, fk, fv);
    flash_attn<<<dim3(16, 16), 256, 0, stream>>>(fq, fk, fv, fo);
    out_proj<<<dim3(1024, 8), 256, 0, stream>>>(fo, wu1, bu1, wu2, bu2, out);
}
